// ScaledDotAttention_28321014350011
// MI455X (gfx1250) — matrix operand order repaired and checked by reading, not compiled, not run on hardware
//
#include <hip/hip_runtime.h>
#include <hip/hip_bf16.h>

typedef __bf16 bf16x16 __attribute__((ext_vector_type(16)));
typedef __bf16 bf16x2  __attribute__((ext_vector_type(2)));
typedef float  f32x8   __attribute__((ext_vector_type(8)));

union Frag { bf16x16 v; unsigned int d[8]; };

#define S_ 2048
#define D_ 128
#define BM 128
#define BN 64
#define NWAVE 8

__device__ __forceinline__ unsigned short f2bf(float f) {
    __bf16 h = (__bf16)f;                       // native v_cvt, RTNE
    return __builtin_bit_cast(unsigned short, h);
}
__device__ __forceinline__ unsigned int pack2bf(float a, float b) {
    bf16x2 h; h.x = (__bf16)a; h.y = (__bf16)b; // lets clang emit packed cvt
    return __builtin_bit_cast(unsigned int, h);
}

__global__ __launch_bounds__(256, 1)
void attn_fwd(const float* __restrict__ Q, const float* __restrict__ K,
              const float* __restrict__ V, float* __restrict__ O)
{
    // LDS staging (80 KB total; 320KB/WGP -> 4 blocks/WGP)
    __shared__ __align__(16) unsigned int   qs4[128 * 64];     // Q*ln2scale bf16 [m][k/2] 32KB
    __shared__ __align__(16) unsigned int   ks4[64 * 64];      // K bf16 [n][k/2]          16KB
    __shared__ __align__(16) unsigned short vts[128 * 64];     // V bf16 transposed [d][n] 16KB
    __shared__ __align__(16) unsigned short ps [NWAVE*16*64];  // per-wave exp(S) [w][m][k]16KB

    const int tid  = threadIdx.x;
    const int wave = tid >> 5;
    const int lane = tid & 31;
    const int half = lane >> 4;
    const int ln   = lane & 15;

    const int qt = blockIdx.x & 15;       // S/BM = 16 query tiles
    const int b  = blockIdx.x >> 4;
    const int q0 = qt * BM;

    // fold softmax scale AND log2(e) into Q so exp(x*scale) == exp2(score)
    const float qscale = 0.08838834764831845f * 1.4426950408889634f;

    const float* qg = Q + ((size_t)b * S_ + q0) * D_;
    const float* kg = K + (size_t)b * S_ * D_;
    const float* vg = V + (size_t)b * S_ * D_;

    // ---- stage Q tile once (fp32 -> pre-scaled bf16) ----
    for (int idx = tid; idx < 128 * 64; idx += 256) {
        int r = idx >> 6, j = idx & 63;
        float2 f = *(const float2*)(qg + r * D_ + j * 2);
        qs4[r * 64 + j] = pack2bf(f.x * qscale, f.y * qscale);
    }
    __syncthreads();

    // ---- Q A-fragments persist for the whole kernel (4 chunks of K=32) ----
    Frag aQ[4];
    {
        int rowq = (wave * 16 + ln) * 64;
        #pragma unroll
        for (int kk = 0; kk < 4; ++kk) {
            #pragma unroll
            for (int g = 0; g < 8; ++g) {
                int kb2 = half * 8 + g;  // k_base/2
                aQ[kk].d[g] = qs4[rowq + kk * 16 + kb2];
            }
        }
    }

    f32x8 oacc[8];
    #pragma unroll
    for (int t = 0; t < 8; ++t) oacc[t] = {};
    float lsum[8];
    #pragma unroll
    for (int r = 0; r < 8; ++r) lsum[r] = 0.f;

    unsigned int* vt4 = (unsigned int*)vts;
    unsigned int* psu = (unsigned int*)ps;

    for (int kt = 0; kt < S_ / BN; ++kt) {
        __syncthreads();                      // previous K/V tile fully consumed
        const float* kgt = kg + (size_t)kt * BN * D_;
        const float* vgt = vg + (size_t)kt * BN * D_;
        for (int idx = tid; idx < 64 * 64; idx += 256) {
            int r = idx >> 6, j = idx & 63;
            float2 f = *(const float2*)(kgt + r * D_ + j * 2);
            ks4[r * 64 + j] = pack2bf(f.x, f.y);
            float2 g2 = *(const float2*)(vgt + r * D_ + j * 2);
            vts[(j * 2    ) * 64 + r] = f2bf(g2.x);    // transpose: [head][key]
            vts[(j * 2 + 1) * 64 + r] = f2bf(g2.y);
        }
        if (kt + 1 < S_ / BN) {               // hint next tile into cache
            __builtin_prefetch(kgt + BN * D_ + tid * 32, 0, 0);
            __builtin_prefetch(vgt + BN * D_ + tid * 32, 0, 0);
        }
        __syncthreads();

        // ---- S = (Q*s) K^T (wave: 16 rows x 64 keys), exp2 + rowsum + stash P ----
        float lpart[8];
        #pragma unroll
        for (int r = 0; r < 8; ++r) lpart[r] = 0.f;

        #pragma unroll
        for (int t = 0; t < 4; ++t) {
            f32x8 acc = {};
            #pragma unroll
            for (int kk = 0; kk < 4; ++kk) {
                Frag bK;
                int rowb = (t * 16 + ln) * 64 + kk * 16 + half * 8;
                #pragma unroll
                for (int g = 0; g < 8; ++g) bK.d[g] = ks4[rowb + g];
                acc = __builtin_amdgcn_wmma_f32_16x16x32_bf16(
                        false, aQ[kk].v, false, bK.v, (short)0, acc, false, false);
            }
            #pragma unroll
            for (int r = 0; r < 8; ++r) {
                float e = __builtin_amdgcn_exp2f(acc[r]); // exp(score/sqrt(d)) via v_exp_f32
                ps[wave * 1024 + (r + 8 * half) * 64 + t * 16 + ln] = f2bf(e);
                lpart[r] += e;
            }
        }
        // butterfly rowsum across the 16 lanes of each half (masks stay in-half)
        #pragma unroll
        for (int r = 0; r < 8; ++r) {
            float s = lpart[r];
            s += __shfl_xor(s, 1, 32);
            s += __shfl_xor(s, 2, 32);
            s += __shfl_xor(s, 4, 32);
            s += __shfl_xor(s, 8, 32);
            lsum[r] += s;
        }

        __builtin_amdgcn_wave_barrier();      // same-wave LDS ops are HW-ordered

        // ---- O += P V : read P back as A-fragments, V as B-fragments ----
        Frag aP[2];
        #pragma unroll
        for (int c = 0; c < 2; ++c) {
            int base = wave * 512 + ln * 32 + c * 16;
            #pragma unroll
            for (int g = 0; g < 8; ++g) {
                int kb2 = half * 8 + g;
                aP[c].d[g] = psu[base + kb2];
            }
        }
        #pragma unroll
        for (int t2 = 0; t2 < 8; ++t2) {
            #pragma unroll
            for (int c = 0; c < 2; ++c) {
                Frag bV;
                int baseb = (t2 * 16 + ln) * 32 + c * 16 + half * 8;
                #pragma unroll
                for (int g = 0; g < 8; ++g) bV.d[g] = vt4[baseb + g];
                oacc[t2] = __builtin_amdgcn_wmma_f32_16x16x32_bf16(
                             false, aP[c].v, false, bV.v, (short)0, oacc[t2], false, false);
            }
        }
    }

    // ---- normalize (rcp instead of 64 divides) and store fp32 ----
    float inv[8];
    #pragma unroll
    for (int r = 0; r < 8; ++r) inv[r] = __builtin_amdgcn_rcpf(lsum[r]);

    float* og = O + ((size_t)b * S_ + q0) * D_;
    #pragma unroll
    for (int t2 = 0; t2 < 8; ++t2) {
        #pragma unroll
        for (int r = 0; r < 8; ++r) {
            int row = wave * 16 + r + 8 * half;
            int col = t2 * 16 + ln;
            og[row * D_ + col] = oacc[t2][r] * inv[r];
        }
    }
}

extern "C" void kernel_launch(void* const* d_in, const int* in_sizes, int n_in,
                              void* d_out, int out_size, void* d_ws, size_t ws_size,
                              hipStream_t stream) {
    const float* q = (const float*)d_in[0];
    const float* k = (const float*)d_in[1];
    const float* v = (const float*)d_in[2];
    float* o = (float*)d_out;
    dim3 grid(32 * (S_ / BM));   // B * query-tiles = 512 workgroups
    dim3 block(256);             // 8 waves (wave32)
    hipLaunchKernelGGL(attn_fwd, grid, block, 0, stream, q, k, v, o);
}
